// CrossAttention_83949430768095
// MI455X (gfx1250) — matrix operand order repaired and checked by reading, not compiled, not run on hardware
//
#include <hip/hip_runtime.h>
#include <hip/hip_bf16.h>

// ---------------------------------------------------------------------------
// CDNA5 (gfx1250) patch-wise cross attention.
// B=16, C=PN=64, H=W=256, kh=kw=32, L=1024, PH=PW=8.
// All matmuls but the Q projection via v_wmma_f32_16x16x32_bf16 (wave32).
// WMMA operand reads from LDS are 16-contiguous-bf16 (2x ds_load_b128).
// Biases are folded into the WMMA C-operand initial value.
// C tiles are restaged via per-wave LDS so global stores are b128 rows.
// attn kernel streams qp/K chunks with async global->LDS + double buffering.
// ---------------------------------------------------------------------------

typedef __bf16 bf16_t;
typedef __attribute__((ext_vector_type(16))) __bf16 v16bf;
typedef __attribute__((ext_vector_type(8)))  float  v8f;
typedef int v4i_vs __attribute__((vector_size(16)));  // matches async builtin

#if defined(__has_builtin)
#if __has_builtin(__builtin_amdgcn_global_load_async_to_lds_b128)
#define HAVE_ASYNC 1
#endif
#endif
#ifndef HAVE_ASYNC
#define HAVE_ASYNC 0
#endif

__device__ __forceinline__ bf16_t f2bf(float f) {
  __hip_bfloat16 h = __float2bfloat16(f);
  return *reinterpret_cast<bf16_t*>(&h);
}

__device__ __forceinline__ v8f wmma_bf16(v16bf a, v16bf b, v8f c) {
  // (neg_a, A, neg_b, B, c_mod, C, reuse_a, reuse_b)
  return __builtin_amdgcn_wmma_f32_16x16x32_bf16(false, a, false, b, (short)0, c,
                                                 false, false);
}

// Async 16-byte global->LDS copy (per-lane addresses), fallback: uint4 copy.
__device__ __forceinline__ void cp_async16(const bf16_t* g, bf16_t* l) {
#if HAVE_ASYNC
  __builtin_amdgcn_global_load_async_to_lds_b128(
      (__attribute__((address_space(1))) v4i_vs*)(unsigned long long)(size_t)g,
      (__attribute__((address_space(3))) v4i_vs*)(unsigned)(size_t)l, 0, 0);
#else
  *reinterpret_cast<uint4*>(l) = *reinterpret_cast<const uint4*>(g);
#endif
}

__device__ __forceinline__ void async_fence() {
#if HAVE_ASYNC
#if __has_builtin(__builtin_amdgcn_s_wait_asynccnt)
  __builtin_amdgcn_s_wait_asynccnt(0);
#else
  asm volatile("s_wait_asynccnt 0" ::: "memory");
#endif
#endif
}

// A tile (16x32 bf16, M x K) from LDS laid out [m][k], row stride RS.
template <int RS>
__device__ __forceinline__ v16bf load_A(const bf16_t* base, int m0, int k0,
                                        int lr, int half) {
  v16bf a;
  const bf16_t* row = base + (m0 + lr) * RS + k0 + half * 16;
#pragma unroll
  for (int e = 0; e < 16; ++e) a[e] = row[e];
  return a;
}

// B tile (32x16 bf16, K x N) from LDS laid out [n][k] (transposed), stride RS.
template <int RS>
__device__ __forceinline__ v16bf load_B_nk(const bf16_t* base, int k0, int n0,
                                           int lr, int half) {
  v16bf b;
  const bf16_t* row = base + (n0 + lr) * RS + k0 + half * 16;
#pragma unroll
  for (int e = 0; e < 16; ++e) b[e] = row[e];
  return b;
}

// ---------------------------------------------------------------------------
// Kernel 1: Q projection (tiny). qp[b][p][l] bf16. grid (4,16) x 256.
// ---------------------------------------------------------------------------
__global__ void qproj_kernel(const float* __restrict__ q,
                             const float* __restrict__ Wq,
                             const float* __restrict__ bq,
                             bf16_t* __restrict__ qp) {
  __shared__ float Wq_s[64 * 64];
  __shared__ float bq_s[64];
  int t = threadIdx.x;
  for (int idx = t; idx < 64 * 64; idx += 256) Wq_s[idx] = Wq[idx];
  if (t < 64) bq_s[t] = bq[t];
  __syncthreads();
  int b = blockIdx.y;
  int l = blockIdx.x * 256 + t;
  float qv[64];
#pragma unroll
  for (int d = 0; d < 64; ++d) qv[d] = q[((size_t)b * 64 + d) * 1024 + l];
  for (int p = 0; p < 64; ++p) {
    float s = bq_s[p];
#pragma unroll
    for (int d = 0; d < 64; ++d) s += Wq_s[p * 64 + d] * qv[d];
    qp[((size_t)b * 64 + p) * 1024 + l] = f2bf(s);
  }
}

// ---------------------------------------------------------------------------
// Kernel 2: fused K/V 1x1 conv + cross-patch rearrange + bias + abs-pos.
// grid (256,16) x 256 threads (8 waves); wave w = patch column pc.
// lf staged transposed [x][d] (RS=72); bias+abs folded into C-init; C tiles
// restaged into (dead) lf_s[w] and bulk-stored as b128 rows.
// ---------------------------------------------------------------------------
__global__ void kvproj_kernel(const float* __restrict__ lf,
                              const float* __restrict__ Wk,
                              const float* __restrict__ bk,
                              const float* __restrict__ Wv,
                              const float* __restrict__ bv,
                              const float* __restrict__ abs_k,
                              const float* __restrict__ abs_v,
                              bf16_t* __restrict__ kc,
                              bf16_t* __restrict__ vc) {
  __shared__ alignas(16) bf16_t Wk_s[64 * 64];
  __shared__ alignas(16) bf16_t Wv_s[64 * 64];
  __shared__ alignas(16) bf16_t lf_s[8][32 * 72];  // [x][d]; reused as C stage
  __shared__ alignas(16) float bias2_s[2][8][64];  // b[c] + abs[c][q], per wave
  int t = threadIdx.x;
  int b = blockIdx.y, h = blockIdx.x;
  for (int idx = t; idx < 64 * 64; idx += 256) {
    Wk_s[idx] = f2bf(Wk[idx]);
    Wv_s[idx] = f2bf(Wv[idx]);
  }
  int w = t >> 5, lane = t & 31;
  int pc = w;
  int pr = h >> 5, y = h & 31;
  int qpatch = pr * 8 + pc;

  // Combined bias for this wave's patch q: 8 global loads per thread.
#pragma unroll
  for (int mat = 0; mat < 2; ++mat) {
    const float* absm = mat ? abs_v : abs_k;
    const float* bbg = mat ? bv : bk;
    for (int c = lane; c < 64; c += 32)
      bias2_s[mat][w][c] = bbg[c] + absm[c * 64 + qpatch];
  }

  // Stage lf tile [64 in-ch x 32 pixels] f32 -> bf16, transposed [x][d].
  const float* src = lf + ((size_t)b * 64 * 256 + h) * 256 + pc * 32;
  for (int idx = lane; idx < 512; idx += 32) {
    int d = idx >> 3, seg = idx & 7;
    float4 v = *reinterpret_cast<const float4*>(src + (size_t)d * 65536 + seg * 4);
    int x = seg * 4;
    lf_s[w][(x + 0) * 72 + d] = f2bf(v.x);
    lf_s[w][(x + 1) * 72 + d] = f2bf(v.y);
    lf_s[w][(x + 2) * 72 + d] = f2bf(v.z);
    lf_s[w][(x + 3) * 72 + d] = f2bf(v.w);
  }
  __syncthreads();

  int lr = lane & 15, half = lane >> 4;

  // Hoist the 4 B tiles; lf_s[w] is dead afterwards (per-wave DS ordering).
  v16bf btile[2][2];
#pragma unroll
  for (int ks = 0; ks < 2; ++ks)
#pragma unroll
    for (int nt = 0; nt < 2; ++nt)
      btile[ks][nt] = load_B_nk<72>(lf_s[w], ks * 32, nt * 16, lr, half);

  bf16_t* stg = lf_s[w];  // reuse as [c][x] stage (64 x 32 bf16 = 4KB)

#pragma unroll
  for (int mat = 0; mat < 2; ++mat) {
    const bf16_t* Ws = mat ? Wv_s : Wk_s;
    bf16_t* outp = mat ? vc : kc;
#pragma unroll
    for (int mt = 0; mt < 4; ++mt) {
      v8f binit;
#pragma unroll
      for (int r = 0; r < 8; ++r)
        binit[r] = bias2_s[mat][w][mt * 16 + half * 8 + r];
#pragma unroll
      for (int nt = 0; nt < 2; ++nt) {
        v8f acc = binit;  // bias folded into C operand
#pragma unroll
        for (int ks = 0; ks < 2; ++ks) {
          v16bf a = load_A<64>(Ws, mt * 16, ks * 32, lr, half);
          acc = wmma_bf16(a, btile[ks][nt], acc);
        }
#pragma unroll
        for (int r = 0; r < 8; ++r)
          stg[(mt * 16 + half * 8 + r) * 32 + nt * 16 + lr] = f2bf(acc[r]);
      }
    }
    // Bulk store: 64 rows (c) x 32 bf16 (l) as b128 segments.
    bf16_t* dst = outp + ((size_t)(b * 64) * 64 + qpatch) * 1024 + (size_t)y * 32;
#pragma unroll
    for (int j = 0; j < 8; ++j) {
      int idx = lane + 32 * j;  // 0..255
      int c = idx >> 2, seg = idx & 3;
      *reinterpret_cast<uint4*>(dst + (size_t)c * 65536 + seg * 8) =
          *reinterpret_cast<const uint4*>(&stg[c * 32 + seg * 8]);
    }
  }
}

// ---------------------------------------------------------------------------
// Kernel 3: per-(b,i) attention. grid (64,16) x 128 threads (4 waves).
// matmul1 streams qp/K l-chunks via async-to-LDS with double buffering.
// matmul2 stages V transposed [l][q]; O tiles restaged via (dead) qs and
// bulk-stored as b128 rows.
// ---------------------------------------------------------------------------
__global__ void attn_kernel(const bf16_t* __restrict__ qp,
                            const bf16_t* __restrict__ kc,
                            const bf16_t* __restrict__ vc,
                            bf16_t* __restrict__ O) {
  __shared__ alignas(16) bf16_t qs[2][64 * 32];   // also O stage in matmul2
  __shared__ alignas(16) bf16_t ks_[2][64 * 32];
  __shared__ alignas(16) float at[64 * 64];
  __shared__ alignas(16) bf16_t at_bf[64 * 64];
  __shared__ alignas(16) bf16_t vs[64 * 72];      // [l][q], padded
  int t = threadIdx.x;  // 0..127
  int b = blockIdx.y, i = blockIdx.x;
  int w = t >> 5, lane = t & 31, lr = lane & 15, half = lane >> 4;
  const bf16_t* qpb = qp + (size_t)b * 64 * 1024;
  const bf16_t* kci = kc + (size_t)(b * 64 + i) * 64 * 1024;
  const bf16_t* vci = vc + (size_t)(b * 64 + i) * 64 * 1024;
  bf16_t* Oi = O + (size_t)(b * 64 + i) * 64 * 1024;

  auto stage_qk = [&](int lc, int buf) {
    int l0 = lc * 32;
    for (int idx = t; idx < 256; idx += 128) {
      int row = idx >> 2, seg = idx & 3;
      cp_async16(qpb + (size_t)row * 1024 + l0 + seg * 8,
                 &qs[buf][row * 32 + seg * 8]);
      cp_async16(kci + (size_t)row * 1024 + l0 + seg * 8,
                 &ks_[buf][row * 32 + seg * 8]);
    }
  };

  // ---- matmul1: attn[p,q] = sum_l qp[p,l] * K[q,l] ----
  v8f acc[4] = {};
  stage_qk(0, 0);
  async_fence();
  __syncthreads();
  for (int lc = 0; lc < 32; ++lc) {
    int cur = lc & 1;
    if (lc + 1 < 32) stage_qk(lc + 1, cur ^ 1);  // overlapped prefetch
    v16bf a = load_A<32>(qs[cur], w * 16, 0, lr, half);
#pragma unroll
    for (int nt = 0; nt < 4; ++nt) {
      v16bf bm = load_B_nk<32>(ks_[cur], 0, nt * 16, lr, half);
      acc[nt] = wmma_bf16(a, bm, acc[nt]);
    }
    async_fence();
    __syncthreads();
  }
#pragma unroll
  for (int nt = 0; nt < 4; ++nt)
#pragma unroll
    for (int r = 0; r < 8; ++r)
      at[(w * 16 + half * 8 + r) * 64 + nt * 16 + lr] = acc[nt][r];
  __syncthreads();

  // ---- softmax over q (rows of at) ----
  if (t < 64) {
    float m = -1e30f;
    for (int qx = 0; qx < 64; ++qx) m = fmaxf(m, at[t * 64 + qx]);
    float s = 0.f;
    for (int qx = 0; qx < 64; ++qx) s += __expf(at[t * 64 + qx] - m);
    float inv = 1.f / s;
    for (int qx = 0; qx < 64; ++qx)
      at_bf[t * 64 + qx] = f2bf(__expf(at[t * 64 + qx] - m) * inv);
  }
  __syncthreads();

  // ---- matmul2: O[p,l] = sum_q attn[p,q] * V[q,l] ----
  bf16_t* stg = &qs[0][0] + w * 1024;  // per-wave [p][16 l] stage (2KB)
  for (int ch = 0; ch < 16; ++ch) {
    int l0 = ch * 64;
    for (int idx = t; idx < 512; idx += 128) {
      int rq = idx >> 3, seg = idx & 7;  // 8 consecutive l, fixed q
      uint4 u = *reinterpret_cast<const uint4*>(vci + (size_t)rq * 1024 + l0 +
                                                seg * 8);
      const bf16_t* pv = reinterpret_cast<const bf16_t*>(&u);
      int lb = seg * 8;
#pragma unroll
      for (int j = 0; j < 8; ++j) vs[(lb + j) * 72 + rq] = pv[j];
    }
    __syncthreads();
    int n0 = w * 16;
    v16bf b0 = load_B_nk<72>(vs, 0, n0, lr, half);
    v16bf b1 = load_B_nk<72>(vs, 32, n0, lr, half);
#pragma unroll
    for (int mt = 0; mt < 4; ++mt) {
      v8f oacc = {};
      oacc = wmma_bf16(load_A<64>(at_bf, mt * 16, 0, lr, half), b0, oacc);
      oacc = wmma_bf16(load_A<64>(at_bf, mt * 16, 32, lr, half), b1, oacc);
#pragma unroll
      for (int r = 0; r < 8; ++r)
        stg[(mt * 16 + half * 8 + r) * 16 + lr] = f2bf(oacc[r]);
    }
    // Bulk store: 64 rows (p) x 16 bf16 as b128 segments.
#pragma unroll
    for (int j = 0; j < 4; ++j) {
      int idx = lane + 32 * j;  // 0..127
      int p = idx >> 1, seg = idx & 1;
      *reinterpret_cast<uint4*>(Oi + (size_t)p * 1024 + l0 + n0 + seg * 8) =
          *reinterpret_cast<const uint4*>(&stg[p * 16 + seg * 8]);
    }
    __syncthreads();
  }
}

// ---------------------------------------------------------------------------
// Kernel 4: output 1x1 conv + fold back to image. grid (256,16) x 256 thr.
// O chunk staged transposed [x][i]; bo folded into C-init; f32 output
// restaged via (dead) Os[w] and bulk-stored as b128 rows.
// ---------------------------------------------------------------------------
__global__ void outconv_kernel(const bf16_t* __restrict__ O,
                               const float* __restrict__ Wo,
                               const float* __restrict__ bo,
                               float* __restrict__ out) {
  __shared__ alignas(16) bf16_t Wo_s[64 * 64];
  __shared__ alignas(16) bf16_t Os[8][32 * 72];  // [x][i]; reused as f32 stage
  __shared__ alignas(16) float bo_s[64];
  int t = threadIdx.x;
  int b = blockIdx.y, h = blockIdx.x;
  for (int idx = t; idx < 4096; idx += 256) Wo_s[idx] = f2bf(Wo[idx]);
  if (t < 64) bo_s[t] = bo[t];
  int w = t >> 5, lane = t & 31, lr = lane & 15, half = lane >> 4;
  int pc = w, pr = h >> 5, y = h & 31;
  int p = pr * 8 + pc;
  const bf16_t* src = O + ((size_t)b * 64 * 64 + p) * 1024 + (size_t)y * 32;
  for (int idx = lane; idx < 256; idx += 32) {
    int i = idx >> 2, seg = idx & 3;
    uint4 u = *reinterpret_cast<const uint4*>(src + (size_t)i * 65536 + seg * 8);
    const bf16_t* pv = reinterpret_cast<const bf16_t*>(&u);
    int xb = seg * 8;
#pragma unroll
    for (int j = 0; j < 8; ++j) Os[w][(xb + j) * 72 + i] = pv[j];
  }
  __syncthreads();

  v16bf btile[2][2];
#pragma unroll
  for (int ks = 0; ks < 2; ++ks)
#pragma unroll
    for (int nt = 0; nt < 2; ++nt)
      btile[ks][nt] = load_B_nk<72>(Os[w], ks * 32, nt * 16, lr, half);

  float* stg = reinterpret_cast<float*>(Os[w]);  // [16 o][32 x] f32 = 2KB
  float* outp = out + ((size_t)b * 64 * 256 + h) * 256 + pc * 32;
#pragma unroll
  for (int mt = 0; mt < 4; ++mt) {
    v8f binit;
#pragma unroll
    for (int r = 0; r < 8; ++r) binit[r] = bo_s[mt * 16 + half * 8 + r];
#pragma unroll
    for (int nt = 0; nt < 2; ++nt) {
      v8f acc = binit;  // bo folded into C operand
#pragma unroll
      for (int ks = 0; ks < 2; ++ks) {
        v16bf a = load_A<64>(Wo_s, mt * 16, ks * 32, lr, half);
        acc = wmma_bf16(a, btile[ks][nt], acc);
      }
#pragma unroll
      for (int r = 0; r < 8; ++r)
        stg[(half * 8 + r) * 32 + nt * 16 + lr] = acc[r];
    }
    // Bulk store: 16 rows (o) x 32 f32 as b128 segments.
#pragma unroll
    for (int j = 0; j < 4; ++j) {
      int idx = lane + 32 * j;  // 0..127
      int orow = idx >> 3, seg = idx & 7;
      *reinterpret_cast<float4*>(outp + (size_t)(mt * 16 + orow) * 65536 +
                                 seg * 4) =
          *reinterpret_cast<const float4*>(&stg[orow * 32 + seg * 4]);
    }
  }
}

// ---------------------------------------------------------------------------
extern "C" void kernel_launch(void* const* d_in, const int* in_sizes, int n_in,
                              void* d_out, int out_size, void* d_ws,
                              size_t ws_size, hipStream_t stream) {
  (void)in_sizes; (void)n_in; (void)out_size; (void)ws_size;
  const float* q    = (const float*)d_in[0];
  const float* lf   = (const float*)d_in[1];
  const float* Wq   = (const float*)d_in[2];
  const float* bq   = (const float*)d_in[3];
  const float* Wk   = (const float*)d_in[4];
  const float* bk   = (const float*)d_in[5];
  const float* Wv   = (const float*)d_in[6];
  const float* bv   = (const float*)d_in[7];
  const float* absk = (const float*)d_in[8];
  const float* absv = (const float*)d_in[9];
  const float* Wo   = (const float*)d_in[10];
  const float* bo   = (const float*)d_in[11];
  float* out = (float*)d_out;

  // Workspace: qp 2MB | kc 128MB | vc 128MB | O 128MB (all bf16)
  char* ws = (char*)d_ws;
  const size_t KV_BYTES = 134217728ull;  // 16*64*64*1024*2
  bf16_t* qp = (bf16_t*)(ws);
  bf16_t* kc = (bf16_t*)(ws + (2ull << 20));
  bf16_t* vc = (bf16_t*)(ws + (2ull << 20) + KV_BYTES);
  bf16_t* Ob = (bf16_t*)(ws + (2ull << 20) + 2 * KV_BYTES);

  qproj_kernel<<<dim3(4, 16), 256, 0, stream>>>(q, Wq, bq, qp);
  kvproj_kernel<<<dim3(256, 16), 256, 0, stream>>>(lf, Wk, bk, Wv, bv, absk,
                                                   absv, kc, vc);
  attn_kernel<<<dim3(64, 16), 128, 0, stream>>>(qp, kc, vc, Ob);
  outconv_kernel<<<dim3(256, 16), 256, 0, stream>>>(Ob, Wo, bo, out);
}
